// CausalSelfAttention_56624848830936
// MI455X (gfx1250) — matrix operand order repaired and checked by reading, not compiled, not run on hardware
//
#include <hip/hip_runtime.h>
#include <hip/hip_bf16.h>

typedef __attribute__((ext_vector_type(4)))  _Float16 v4h;
typedef __attribute__((ext_vector_type(8)))  _Float16 v8h;
typedef __attribute__((ext_vector_type(16))) _Float16 v16h;
typedef __attribute__((ext_vector_type(8)))  float    v8f;

#define B_  2
#define T_  2048
#define D_  1024
#define H_  16
#define DH_ 64
#define BT_ (B_ * T_)          // 4096 tokens
#define N3_ (3 * D_)           // 3072

// ---------------------------------------------------------------------------
// Fragment helpers (CDNA5 wave32 WMMA f16 layouts, CDNA5 ISA)
// A 16x32 f16: lane m = l%16; elems 0..7  -> K = (l/16)*8 + e
//                               elems 8..15 -> K = 16 + (l/16)*8 + (e-8)
// B 32x16 f16: lane n = l%16; same K order as A (elems 0..7 -> (l/16)*8+e, 8..15 -> 16+(l/16)*8+(e-8))
// C/D 16x16 f32: lane l, vgpr r -> row = r + 8*(l/16), col = l%16
// ---------------------------------------------------------------------------
static __device__ __forceinline__ v16h load_fragA(const _Float16* p) {
  v8h lo = *(const v8h*)(p);
  v8h hi = *(const v8h*)(p + 16);
  return __builtin_shufflevector(lo, hi, 0,1,2,3,4,5,6,7,8,9,10,11,12,13,14,15);
}
static __device__ __forceinline__ v16h load_fragB(const _Float16* p) {
  v8h lo = *(const v8h*)(p);
  v8h hi = *(const v8h*)(p + 16);
  return __builtin_shufflevector(lo, hi, 0,1,2,3,4,5,6,7,8,9,10,11,12,13,14,15);
}
static __device__ __forceinline__ v8f wmma_f16(v16h a, v16h b, v8f c) {
  return __builtin_amdgcn_wmma_f32_16x16x32_f16(false, a, false, b, (short)0, c,
                                                false, false);
}

// ---------------------------------------------------------------------------
// f32 -> f16 elementwise (vectorized x4)
// ---------------------------------------------------------------------------
__global__ void k_f32_to_f16(const float* __restrict__ in,
                             _Float16* __restrict__ out, int n4) {
  int i = blockIdx.x * blockDim.x + threadIdx.x;
  if (i < n4) {
    float4 v = ((const float4*)in)[i];
    v4h o;
    o[0] = (_Float16)v.x; o[1] = (_Float16)v.y;
    o[2] = (_Float16)v.z; o[3] = (_Float16)v.w;
    ((v4h*)out)[i] = o;
  }
}

// ---------------------------------------------------------------------------
// fp32 [K][N] -> f16 [N][K]  (weight transpose for B-fragment loads)
// ---------------------------------------------------------------------------
__global__ void k_transpose_f32_f16(const float* __restrict__ in,
                                    _Float16* __restrict__ out,
                                    int K, int N) {
  __shared__ float tile[32][33];
  const int n0 = blockIdx.x * 32;
  const int k0 = blockIdx.y * 32;
  const int tx = threadIdx.x;        // 0..31
  for (int i = threadIdx.y; i < 32; i += 8)
    tile[i][tx] = in[(size_t)(k0 + i) * N + (n0 + tx)];
  __syncthreads();
  for (int i = threadIdx.y; i < 32; i += 8)
    out[(size_t)(n0 + i) * K + (k0 + tx)] = (_Float16)tile[tx][i];
}

// ---------------------------------------------------------------------------
// QKV GEMM: Xh[4096,1024] (f16, row-major) @ WqkvT[3072,1024] (f16, [N][K])
// Block = 128x128 (8 waves, 2x4), wave = 64x32 (4x2 WMMA tiles).
// Epilogue scatters into Qh/Kh [B,H,T,64] and Vt [B,H,64,T].
// ---------------------------------------------------------------------------
static __device__ __forceinline__ void store_qkv_tile(
    const v8f c, int mbase, int nbase, int lm, int lg,
    _Float16* __restrict__ Qh, _Float16* __restrict__ Kh,
    _Float16* __restrict__ Vt) {
  const int n    = nbase + lm;
  const int part = n >> 10;          // 0=Q 1=K 2=V (wave-uniform)
  const int f    = n & 1023;
  const int h    = f >> 6;
  const int dh   = f & 63;
  if (part == 2) {
#pragma unroll
    for (int r = 0; r < 8; ++r) {
      const int m = mbase + r + 8 * lg;
      const int b = m >> 11, t = m & 2047;
      Vt[(((b << 4) + h) * 64 + dh) * T_ + t] = (_Float16)c[r];
    }
  } else {
    _Float16* P = (part == 0) ? Qh : Kh;
#pragma unroll
    for (int r = 0; r < 8; ++r) {
      const int m = mbase + r + 8 * lg;
      const int b = m >> 11, t = m & 2047;
      P[((((b << 4) + h) * T_) + t) * 64 + dh] = (_Float16)c[r];
    }
  }
}

__global__ __launch_bounds__(256) void k_gemm_qkv(
    const _Float16* __restrict__ Xh, const _Float16* __restrict__ WT,
    _Float16* __restrict__ Qh, _Float16* __restrict__ Kh,
    _Float16* __restrict__ Vt) {
  const int lane = threadIdx.x & 31;
  const int wave = threadIdx.x >> 5;
  const int lm = lane & 15, lg = lane >> 4;
  const int m0 = blockIdx.x * 128 + (wave >> 2) * 64;   // 2 wave-rows of 64
  const int n0 = blockIdx.y * 128 + (wave & 3) * 32;    // 4 wave-cols of 32

  const _Float16* ap = Xh + (size_t)(m0 + lm) * D_ + lg * 8;
  const _Float16* bp = WT + (size_t)(n0 + lm) * D_ + lg * 8;

  v8f c[4][2];
#pragma unroll
  for (int i = 0; i < 4; ++i)
#pragma unroll
    for (int j = 0; j < 2; ++j) c[i][j] = (v8f){};

#pragma unroll 2
  for (int k = 0; k < D_; k += 32) {
    v16h a[4], b[2];
#pragma unroll
    for (int i = 0; i < 4; ++i) a[i] = load_fragA(ap + (size_t)i * 16 * D_ + k);
#pragma unroll
    for (int j = 0; j < 2; ++j) b[j] = load_fragB(bp + (size_t)j * 16 * D_ + k);
#pragma unroll
    for (int i = 0; i < 4; ++i)
#pragma unroll
      for (int j = 0; j < 2; ++j) c[i][j] = wmma_f16(a[i], b[j], c[i][j]);
  }
#pragma unroll
  for (int i = 0; i < 4; ++i)
#pragma unroll
    for (int j = 0; j < 2; ++j)
      store_qkv_tile(c[i][j], m0 + i * 16, n0 + j * 16, lm, lg, Qh, Kh, Vt);
}

// ---------------------------------------------------------------------------
// Flash-style causal attention.
// Block = 128 threads (4 independent waves); wave handles 16 query rows.
// Streams 64-key tiles: S = Q@K^T (8 WMMA), online softmax (fp32),
// P via wave-private LDS relayout (2 A-frags), O += P@V (8 WMMA vs V^T).
// ---------------------------------------------------------------------------
__global__ __launch_bounds__(128) void k_attn(
    const _Float16* __restrict__ Qh, const _Float16* __restrict__ Kh,
    const _Float16* __restrict__ Vt, _Float16* __restrict__ Oh) {
  __shared__ __align__(16) _Float16 lds[4][16 * 72];  // per-wave 16x64 P (pad 72)

  const int lane = threadIdx.x & 31;
  const int wave = threadIdx.x >> 5;
  const int lm = lane & 15, lg = lane >> 4;

  const int qt = blockIdx.x & 31;      // T/64 = 32 query tiles of 64
  const int bh = blockIdx.x >> 5;      // 0..31
  const int b = bh >> 4, h = bh & 15;

  const _Float16* Qp = Qh + (size_t)bh * T_ * 64;
  const _Float16* Kp = Kh + (size_t)bh * T_ * 64;
  const _Float16* Vp = Vt + (size_t)bh * 64 * T_;

  const int qbase = qt * 64 + wave * 16;

  // Q A-fragments (kdim 0..31 and 32..63), loaded once
  const v16h aq0 = load_fragA(Qp + (size_t)(qbase + lm) * 64 + lg * 8);
  const v16h aq1 = load_fragA(Qp + (size_t)(qbase + lm) * 64 + 32 + lg * 8);

  v8f o[4];
#pragma unroll
  for (int d = 0; d < 4; ++d) o[d] = (v8f){};
  float mrow[8], lrow[8];
#pragma unroll
  for (int r = 0; r < 8; ++r) { mrow[r] = -3.0e38f; lrow[r] = 0.0f; }

  _Float16* myLds = lds[wave];
  const float scale = 0.125f;          // 1/sqrt(64)

  for (int kt = 0; kt < qbase + 16; kt += 64) {
    // ---- S = Q @ K^T for 64 keys (four 16-col tiles) ----
    v8f s[4];
#pragma unroll
    for (int j = 0; j < 4; ++j) {
      const _Float16* kr = Kp + (size_t)(kt + j * 16 + lm) * 64;
      v8f sj = (v8f){};
      sj = wmma_f16(aq0, load_fragB(kr + lg * 8), sj);
      sj = wmma_f16(aq1, load_fragB(kr + 32 + lg * 8), sj);
      s[j] = sj;
    }
    // ---- scale + causal mask ----
#pragma unroll
    for (int r = 0; r < 8; ++r) {
      const int row = qbase + r + 8 * lg;
#pragma unroll
      for (int j = 0; j < 4; ++j) {
        float v = s[j][r] * scale;
        if (kt + j * 16 + lm > row) v = -3.0e38f;
        s[j][r] = v;
      }
    }
    // ---- online softmax (row reductions across 16-lane halves) ----
    float mnew[8], corr[8];
#pragma unroll
    for (int r = 0; r < 8; ++r) {
      float t = fmaxf(fmaxf(s[0][r], s[1][r]), fmaxf(s[2][r], s[3][r]));
      t = fmaxf(t, __shfl_xor(t, 1));
      t = fmaxf(t, __shfl_xor(t, 2));
      t = fmaxf(t, __shfl_xor(t, 4));
      t = fmaxf(t, __shfl_xor(t, 8));
      mnew[r] = fmaxf(mrow[r], t);
      corr[r] = __expf(mrow[r] - mnew[r]);
      mrow[r] = mnew[r];
    }
#pragma unroll
    for (int r = 0; r < 8; ++r) {
      float ts = 0.0f;
#pragma unroll
      for (int j = 0; j < 4; ++j) {
        const float p = __expf(s[j][r] - mnew[r]);
        s[j][r] = p;
        ts += p;
      }
      ts += __shfl_xor(ts, 1);
      ts += __shfl_xor(ts, 2);
      ts += __shfl_xor(ts, 4);
      ts += __shfl_xor(ts, 8);
      lrow[r] = lrow[r] * corr[r] + ts;
    }
#pragma unroll
    for (int d = 0; d < 4; ++d)
#pragma unroll
      for (int r = 0; r < 8; ++r) o[d][r] *= corr[r];
    // ---- C-layout P (16x64) -> A-fragment layout via wave-private LDS ----
#pragma unroll
    for (int r = 0; r < 8; ++r) {
      const int base = (r + 8 * lg) * 72;
#pragma unroll
      for (int j = 0; j < 4; ++j)
        myLds[base + j * 16 + lm] = (_Float16)s[j][r];
    }
    asm volatile("s_wait_dscnt 0" ::: "memory");
    v16h pa0, pa1;
    {
      v8h lo = *(const v8h*)(myLds + lm * 72 + lg * 8);
      v8h hi = *(const v8h*)(myLds + lm * 72 + 16 + lg * 8);
      pa0 = __builtin_shufflevector(lo, hi, 0,1,2,3,4,5,6,7,8,9,10,11,12,13,14,15);
      v8h lo2 = *(const v8h*)(myLds + lm * 72 + 32 + lg * 8);
      v8h hi2 = *(const v8h*)(myLds + lm * 72 + 48 + lg * 8);
      pa1 = __builtin_shufflevector(lo2, hi2, 0,1,2,3,4,5,6,7,8,9,10,11,12,13,14,15);
    }
    // ---- O += P @ V (V stored transposed [dh][t]) ----
#pragma unroll
    for (int d = 0; d < 4; ++d) {
      const _Float16* vb = Vp + (size_t)(d * 16 + lm) * T_;
      o[d] = wmma_f16(pa0, load_fragB(vb + kt + lg * 8), o[d]);
      o[d] = wmma_f16(pa1, load_fragB(vb + kt + 32 + lg * 8), o[d]);
    }
  }
  // ---- normalize and store O as [B,T,D] f16 ----
#pragma unroll
  for (int r = 0; r < 8; ++r) {
    const int t = qbase + r + 8 * lg;
    const float inv = 1.0f / lrow[r];
    _Float16* orow = Oh + ((size_t)(b * T_ + t)) * D_ + h * 64;
#pragma unroll
    for (int d = 0; d < 4; ++d)
      orow[d * 16 + lm] = (_Float16)(o[d][r] * inv);
  }
}

// ---------------------------------------------------------------------------
// Proj GEMM: Oh[4096,1024] (f16) @ WprojT[1024,1024] (f16 [N][K]) -> fp32 out
// Block = 128x128 (8 waves, 2x4), wave = 64x32 (4x2 WMMA tiles).
// ---------------------------------------------------------------------------
__global__ __launch_bounds__(256) void k_gemm_proj(
    const _Float16* __restrict__ Oh, const _Float16* __restrict__ WT,
    float* __restrict__ out) {
  const int lane = threadIdx.x & 31;
  const int wave = threadIdx.x >> 5;
  const int lm = lane & 15, lg = lane >> 4;
  const int m0 = blockIdx.x * 128 + (wave >> 2) * 64;
  const int n0 = blockIdx.y * 128 + (wave & 3) * 32;

  const _Float16* ap = Oh + (size_t)(m0 + lm) * D_ + lg * 8;
  const _Float16* bp = WT + (size_t)(n0 + lm) * D_ + lg * 8;

  v8f c[4][2];
#pragma unroll
  for (int i = 0; i < 4; ++i)
#pragma unroll
    for (int j = 0; j < 2; ++j) c[i][j] = (v8f){};

#pragma unroll 2
  for (int k = 0; k < D_; k += 32) {
    v16h a[4], b[2];
#pragma unroll
    for (int i = 0; i < 4; ++i) a[i] = load_fragA(ap + (size_t)i * 16 * D_ + k);
#pragma unroll
    for (int j = 0; j < 2; ++j) b[j] = load_fragB(bp + (size_t)j * 16 * D_ + k);
#pragma unroll
    for (int i = 0; i < 4; ++i)
#pragma unroll
      for (int j = 0; j < 2; ++j) c[i][j] = wmma_f16(a[i], b[j], c[i][j]);
  }
#pragma unroll
  for (int i = 0; i < 4; ++i)
#pragma unroll
    for (int j = 0; j < 2; ++j)
#pragma unroll
      for (int r = 0; r < 8; ++r) {
        const int m = m0 + i * 16 + r + 8 * lg;
        out[(size_t)m * D_ + n0 + j * 16 + lm] = c[i][j][r];
      }
}

// ---------------------------------------------------------------------------
// Host launcher
// ---------------------------------------------------------------------------
extern "C" void kernel_launch(void* const* d_in, const int* in_sizes, int n_in,
                              void* d_out, int out_size, void* d_ws,
                              size_t ws_size, hipStream_t stream) {
  const float* x     = (const float*)d_in[0];   // [B,T,D]
  const float* Wqkv  = (const float*)d_in[1];   // [D,3D]
  const float* Wproj = (const float*)d_in[2];   // [D,D]
  float* out = (float*)d_out;                   // [B,T,D]

  char* ws = (char*)d_ws;
  size_t off = 0;
  _Float16* Xh     = (_Float16*)(ws + off); off += (size_t)BT_ * D_ * 2;   // 8 MiB
  _Float16* WqkvT  = (_Float16*)(ws + off); off += (size_t)N3_ * D_ * 2;   // 6 MiB
  _Float16* WprojT = (_Float16*)(ws + off); off += (size_t)D_ * D_ * 2;    // 2 MiB
  _Float16* Qh     = (_Float16*)(ws + off); off += (size_t)BT_ * D_ * 2;   // 8 MiB
  _Float16* Kh     = (_Float16*)(ws + off); off += (size_t)BT_ * D_ * 2;   // 8 MiB
  _Float16* Vt     = (_Float16*)(ws + off); off += (size_t)BT_ * D_ * 2;   // 8 MiB
  _Float16* Oh     = (_Float16*)(ws + off); off += (size_t)BT_ * D_ * 2;   // 8 MiB
  (void)ws_size; (void)in_sizes; (void)n_in; (void)out_size;

  // 1) convert x to f16
  {
    const int n4 = (BT_ * D_) / 4;   // 1,048,576
    k_f32_to_f16<<<n4 / 256, 256, 0, stream>>>(x, Xh, n4);
  }
  // 2) transpose+convert weights to f16 [N][K]
  k_transpose_f32_f16<<<dim3(N3_ / 32, D_ / 32), dim3(32, 8), 0, stream>>>(
      Wqkv, WqkvT, D_, N3_);
  k_transpose_f32_f16<<<dim3(D_ / 32, D_ / 32), dim3(32, 8), 0, stream>>>(
      Wproj, WprojT, D_, D_);
  // 3) QKV GEMM -> Q,K [B,H,T,64], V^T [B,H,64,T]
  k_gemm_qkv<<<dim3(BT_ / 128, N3_ / 128), 256, 0, stream>>>(Xh, WqkvT, Qh, Kh, Vt);
  // 4) causal flash attention -> Oh [B,T,D] f16
  k_attn<<<B_ * H_ * (T_ / 64), 128, 0, stream>>>(Qh, Kh, Vt, Oh);
  // 5) projection GEMM -> fp32 out
  k_gemm_proj<<<dim3(BT_ / 128, D_ / 64), 256, 0, stream>>>(Oh, WprojT, out);
}
